// DeepRBFNetwork_13520557048085
// MI455X (gfx1250) — matrix operand order repaired and checked by reading, not compiled, not run on hardware
//
#include <hip/hip_runtime.h>

typedef __attribute__((ext_vector_type(16))) __bf16 bf16x16;
typedef __attribute__((ext_vector_type(8)))  float  v8f;
typedef __attribute__((ext_vector_type(4)))  int    v4i;

#define KCLS 64
#define AS1 __attribute__((address_space(1)))
#define AS3 __attribute__((address_space(3)))

static __device__ __forceinline__ unsigned short f2bf_bits(float f) {
  unsigned u = __builtin_bit_cast(unsigned, f);
  unsigned r = u + 0x7FFFu + ((u >> 16) & 1u);   // round-to-nearest-even
  return (unsigned short)(r >> 16);
}
static __device__ __forceinline__ float bfbits2f(unsigned short s) {
  unsigned u = ((unsigned)s) << 16;
  return __builtin_bit_cast(float, u);
}
static __device__ __forceinline__ __bf16 bits2bf(unsigned short s) {
  return __builtin_bit_cast(__bf16, s);
}

// --- CDNA5 async global->LDS copy (16B per lane), ASYNCcnt-tracked ----------
static __device__ __forceinline__ void async_copy16(void* lds_dst, const void* gsrc) {
#if __has_builtin(__builtin_amdgcn_global_load_async_to_lds_b128)
  void* g = const_cast<void*>(gsrc);
  __builtin_amdgcn_global_load_async_to_lds_b128(
      (AS1 v4i*)g, (AS3 v4i*)lds_dst, 0, 0);
#else
  unsigned ldsa = (unsigned)(unsigned long long)(AS3 void*)lds_dst;
  asm volatile("global_load_async_to_lds_b128 %0, %1, off"
               :: "v"(ldsa), "v"((unsigned long long)gsrc)
               : "memory");
#endif
}
static __device__ __forceinline__ void wait_async0() {
#if __has_builtin(__builtin_amdgcn_s_wait_asynccnt)
  __builtin_amdgcn_s_wait_asynccnt(0);
#else
  asm volatile("s_wait_asynccnt 0x0" ::: "memory");
#endif
}

// ---------------------------------------------------------------------------
// Prep: convert A[k][d][e] (f32) into bf16 hi/lo arrays, pre-permuted into the
// WMMA 16-bit A-operand layout (A-op = A_k^T: M=e, K=d).
// 16x32 16-bit A layout used here (contiguous k): lane L: M = L%16;
//   element i: K = 16*(L/16) + i
// Storage index: (((k*4 + mt)*2 + kc)*32 + lane)*16 + i
// ---------------------------------------------------------------------------
__global__ void prep_A_kernel(const float* __restrict__ A,
                              unsigned short* __restrict__ hi,
                              unsigned short* __restrict__ lo) {
  int t = blockIdx.x * blockDim.x + threadIdx.x;   // 65536 threads, 4 elems each
  float4 f4 = ((const float4*)A)[t];
  float f[4] = {f4.x, f4.y, f4.z, f4.w};
  int base = t * 4;
#pragma unroll
  for (int j = 0; j < 4; ++j) {
    int lin = base + j;
    int e = lin & 63;
    int d = (lin >> 6) & 63;
    int k = lin >> 12;
    int mt = e >> 4;                       // feature tile (M)
    int kc = d >> 5;                       // K chunk of 32
    int kk = d & 31;
    int h  = kk >> 4;                      // lane half select
    int i  = kk & 15;                      // element within v16bf
    int lane = (e & 15) + 16 * h;
    int off = (((k * 4 + mt) * 2 + kc) * 32 + lane) * 16 + i;
    unsigned short hb = f2bf_bits(f[j]);
    unsigned short lb = f2bf_bits(f[j] - bfbits2f(hb));
    hi[off] = hb;
    lo[off] = lb;
  }
}

// ---------------------------------------------------------------------------
// Main kernel: block = 256 threads = 8 waves; wave handles 2 batch tiles
// (32 rows). Block covers 256 rows -> grid = 65536/256 = 256 blocks.
// Per class: A_hi/A_lo/bias tile double-buffered in LDS via async-to-LDS,
// shared by all 8 waves; 48 bf16 WMMAs per wave per class.
// ---------------------------------------------------------------------------
__global__ __launch_bounds__(256) void rbf_main_kernel(
    const float* __restrict__ x, const float* __restrict__ bias,
    const unsigned short* __restrict__ Ahi, const unsigned short* __restrict__ Alo,
    float* __restrict__ out) {
  __shared__ __align__(32) short s_hi[2][4096];
  __shared__ __align__(32) short s_lo[2][4096];
  __shared__ __align__(16) float s_b[2][64];

  const int t = threadIdx.x;
  const int wave = t >> 5;
  const int lane = t & 31;
  const int bn = lane & 15;      // batch row within tile (N)
  const int h  = lane >> 4;      // lane half
  const int rowBase = blockIdx.x * 256 + wave * 32;

  // Build x fragments (B-operand, x^T: K=d, N=batch). 32-row slab, split hi/lo.
  // 16-bit B 32x16 layout used here (contiguous k): lane L: n=L%16; element i: K = (L/16)*16 + i.
  bf16x16 xhi[2][2], xlo[2][2];
#pragma unroll
  for (int bt = 0; bt < 2; ++bt) {
#pragma unroll
    for (int kc = 0; kc < 2; ++kc) {
      const float* p = x + (size_t)(rowBase + bt * 16 + bn) * 64 + kc * 32 + h * 16;
#pragma unroll
      for (int i = 0; i < 16; i += 4) {
        float4 v = *(const float4*)(p + i);
        float vv[4] = {v.x, v.y, v.z, v.w};
#pragma unroll
        for (int j = 0; j < 4; ++j) {
          unsigned short hb = f2bf_bits(vv[j]);
          xhi[bt][kc][i + j] = bits2bf(hb);
          xlo[bt][kc][i + j] = bits2bf(f2bf_bits(vv[j] - bfbits2f(hb)));
        }
      }
    }
  }

  // Async staging of one class tile (hi 8KB + lo 8KB + bias 256B) into buffer.
  // Each of 256 threads moves 2x16B of hi, 2x16B of lo; threads 0..15 do bias.
  auto stage = [&](int kcls, int buf) {
    const unsigned short* gh = Ahi + (size_t)kcls * 4096;
    const unsigned short* gl = Alo + (size_t)kcls * 4096;
    async_copy16(&s_hi[buf][t * 8],        gh + t * 8);
    async_copy16(&s_hi[buf][2048 + t * 8], gh + 2048 + t * 8);
    async_copy16(&s_lo[buf][t * 8],        gl + t * 8);
    async_copy16(&s_lo[buf][2048 + t * 8], gl + 2048 + t * 8);
    if (t < 16) async_copy16(&s_b[buf][t * 4], bias + (size_t)kcls * 64 + t * 4);
  };

  stage(0, 0);
  wait_async0();
  __syncthreads();

  for (int k = 0; k < KCLS; ++k) {
    const int cur = k & 1;
    if (k + 1 < KCLS) stage(k + 1, 1 - cur);   // async L2->LDS for next class

    v8f acc[4][2] = {};
    const short* hh = s_hi[cur];
    const short* ll = s_lo[cur];
#pragma unroll
    for (int mt = 0; mt < 4; ++mt) {
#pragma unroll
      for (int kc = 0; kc < 2; ++kc) {
        bf16x16 Af = *(const bf16x16*)&hh[((mt * 2 + kc) * 32 + lane) * 16];
        bf16x16 Al = *(const bf16x16*)&ll[((mt * 2 + kc) * 32 + lane) * 16];
#pragma unroll
        for (int bt = 0; bt < 2; ++bt) {
          acc[mt][bt] = __builtin_amdgcn_wmma_f32_16x16x32_bf16(
              false, Af, false, xhi[bt][kc], (short)0, acc[mt][bt], false, false);
          acc[mt][bt] = __builtin_amdgcn_wmma_f32_16x16x32_bf16(
              false, Al, false, xhi[bt][kc], (short)0, acc[mt][bt], false, false);
          acc[mt][bt] = __builtin_amdgcn_wmma_f32_16x16x32_bf16(
              false, Af, false, xlo[bt][kc], (short)0, acc[mt][bt], false, false);
        }
      }
    }

    // bias + squared-sum over features; lane holds e = mt*16 + 8*h + r of row bn
#pragma unroll
    for (int bt = 0; bt < 2; ++bt) {
      float s = 0.f;
#pragma unroll
      for (int mt = 0; mt < 4; ++mt) {
        const float* bb = &s_b[cur][mt * 16 + 8 * h];
#pragma unroll
        for (int r = 0; r < 8; ++r) {
          float v = acc[mt][bt][r] + bb[r];
          s = fmaf(v, v, s);
        }
      }
      s += __shfl_xor(s, 16, 32);        // combine the two lane halves (all 64 e)
      float dist = sqrtf(s);
      if (lane < 16)
        out[(size_t)(rowBase + bt * 16 + lane) * 64 + k] = dist;
    }

    wait_async0();     // our wave's async writes into the other buffer done
    __syncthreads();   // everyone's done -> next buffer valid, this one reusable
  }
}

extern "C" void kernel_launch(void* const* d_in, const int* in_sizes, int n_in,
                              void* d_out, int out_size, void* d_ws, size_t ws_size,
                              hipStream_t stream) {
  (void)in_sizes; (void)n_in; (void)out_size; (void)ws_size;
  const float* x = (const float*)d_in[0];   // (65536, 64)
  const float* A = (const float*)d_in[1];   // (64, 64, 64)
  const float* b = (const float*)d_in[2];   // (64, 64)
  float* out = (float*)d_out;               // (65536, 64)

  unsigned short* Ahi = (unsigned short*)d_ws;          // 512 KB
  unsigned short* Alo = Ahi + (size_t)KCLS * 4096;      // 512 KB

  // 64*64*64 f32 elements, 4 per thread -> 65536 threads
  prep_A_kernel<<<256, 256, 0, stream>>>(A, Ahi, Alo);
  // 65536 rows / 256 rows-per-block
  rbf_main_kernel<<<256, 256, 0, stream>>>(x, b, Ahi, Alo, out);
}
